// DiagonalStateElman_66142496358518
// MI455X (gfx1250) — matrix operand order repaired and checked by reading, not compiled, not run on hardware
//
#include <hip/hip_runtime.h>

// ---------------------------------------------------------------------------
// Types for CDNA5 WMMA (wave32): V_WMMA_F32_16X16X32_BF16
// ---------------------------------------------------------------------------
typedef __bf16 bf16;
typedef float  v8f    __attribute__((ext_vector_type(8)));
typedef bf16   v16bf  __attribute__((ext_vector_type(16)));
typedef bf16   bf16x8 __attribute__((ext_vector_type(8)));

#define BATCH 8
#define TLEN  2048
#define DIMD  1024   // dim
#define DINN  1024   // d_inner
#define DST   2048   // d_state
#define NROW  (BATCH * TLEN)   // 16384 (b*T+t rows)

__device__ __forceinline__ float siluf(float v) {
  return v / (1.0f + __expf(-v));
}

__device__ __forceinline__ v16bf frag_zero() {
  v16bf z;
#pragma unroll
  for (int i = 0; i < 16; ++i) z[i] = (bf16)0.0f;
  return z;
}

// A fragment (16x32 bf16, M rows x K): lane holds row m=lane&15,
// chunks K = 8*(lane>>4)..+7 (V0-3) and +16.. (V4-7). Row-major src, ld elems.
__device__ __forceinline__ v16bf load_a(const bf16* base, int ld, int lane) {
  const bf16* p = base + (size_t)(lane & 15) * ld + ((lane >> 4) << 3);
  bf16x8 c0 = *(const bf16x8*)(p);
  bf16x8 c1 = *(const bf16x8*)(p + 16);
  return __builtin_shufflevector(c0, c1, 0,1,2,3,4,5,6,7,8,9,10,11,12,13,14,15);
}

// Same, but only 8 valid rows in memory (batch pad 8->16): rows >=8 are zero.
__device__ __forceinline__ v16bf load_a_pad8(const bf16* base, int ld, int lane) {
  int m = lane & 15;
  if (m >= 8) return frag_zero();
  const bf16* p = base + (size_t)m * ld + ((lane >> 4) << 3);
  bf16x8 c0 = *(const bf16x8*)(p);
  bf16x8 c1 = *(const bf16x8*)(p + 16);
  return __builtin_shufflevector(c0, c1, 0,1,2,3,4,5,6,7,8,9,10,11,12,13,14,15);
}

// B fragment (32x16 bf16, K x N): lane holds col n=lane&15,
// chunks K = 8*(lane>>4)..+7 and +16.. (same as A). Src stored transposed [n][k], ld = K stride.
__device__ __forceinline__ v16bf load_b(const bf16* base, int ld, int lane) {
  const bf16* p = base + (size_t)(lane & 15) * ld + ((lane >> 4) << 3);
  bf16x8 c0 = *(const bf16x8*)(p);
  bf16x8 c1 = *(const bf16x8*)(p + 16);
  return __builtin_shufflevector(c0, c1, 0,1,2,3,4,5,6,7,8,9,10,11,12,13,14,15);
}

__device__ __forceinline__ v8f wmma_bf16(v16bf a, v16bf b, v8f c) {
  return __builtin_amdgcn_wmma_f32_16x16x32_bf16(false, a, false, b, (short)0, c,
                                                 false, false);
}

__device__ __forceinline__ v8f acc_zero() {
  v8f a;
#pragma unroll
  for (int i = 0; i < 8; ++i) a[i] = 0.0f;
  return a;
}

// ---------------------------------------------------------------------------
// Sense-reversing device-wide barrier (counter self-resets -> replay safe)
// ---------------------------------------------------------------------------
__device__ __forceinline__ void grid_barrier(unsigned* cnt, unsigned* gen) {
  __threadfence();
  __syncthreads();
  if (threadIdx.x == 0) {
    unsigned g = __hip_atomic_load(gen, __ATOMIC_RELAXED, __HIP_MEMORY_SCOPE_AGENT);
    unsigned arrived =
        __hip_atomic_fetch_add(cnt, 1u, __ATOMIC_ACQ_REL, __HIP_MEMORY_SCOPE_AGENT);
    if (arrived == gridDim.x - 1) {
      __hip_atomic_store(cnt, 0u, __ATOMIC_RELAXED, __HIP_MEMORY_SCOPE_AGENT);
      __hip_atomic_fetch_add(gen, 1u, __ATOMIC_ACQ_REL, __HIP_MEMORY_SCOPE_AGENT);
    } else {
      while (__hip_atomic_load(gen, __ATOMIC_ACQUIRE, __HIP_MEMORY_SCOPE_AGENT) == g) {
        __builtin_amdgcn_s_sleep(4);
      }
    }
  }
  __syncthreads();
  __threadfence();
}

// ---------------------------------------------------------------------------
// Prep kernels
// ---------------------------------------------------------------------------
__global__ void conv_f2b_kernel(const float* __restrict__ src, bf16* __restrict__ dst,
                                long long n) {
  long long stride = (long long)gridDim.x * blockDim.x;
  for (long long i = (long long)blockIdx.x * blockDim.x + threadIdx.x; i < n; i += stride)
    dst[i] = (bf16)src[i];
}

// dst[n*K + k] = (bf16) src[k*N + n]    (src is [K][N] row-major)
__global__ void transpose_f2b_kernel(const float* __restrict__ src, bf16* __restrict__ dst,
                                     int K, int N) {
  long long total = (long long)K * N;
  long long stride = (long long)gridDim.x * blockDim.x;
  for (long long i = (long long)blockIdx.x * blockDim.x + threadIdx.x; i < total;
       i += stride) {
    int n = (int)(i / K);
    int k = (int)(i % K);
    dst[i] = (bf16)src[(size_t)k * N + n];
  }
}

__global__ void init_state_kernel(const float* __restrict__ h0, float* __restrict__ h32,
                                  unsigned* __restrict__ sync) {
  long long total = 16LL * DST;
  long long stride = (long long)gridDim.x * blockDim.x;
  for (long long i = (long long)blockIdx.x * blockDim.x + threadIdx.x; i < total;
       i += stride) {
    int m = (int)(i / DST);
    h32[i] = (m < BATCH) ? h0[i] : 0.0f;
  }
  if (blockIdx.x == 0 && threadIdx.x == 0) { sync[0] = 0u; sync[1] = 0u; }
}

// ---------------------------------------------------------------------------
// Input projection: xz = x @ W_in^T, fused SiLU split. 2x2 register blocking:
// each wave owns a 32x32 output tile -> 4 WMMAs per 4 fragment loads.
// ---------------------------------------------------------------------------
__device__ __forceinline__ void store_xz_tile(v8f acc, int row0, int col0, int lane,
                                              bf16* __restrict__ u,
                                              float* __restrict__ gz) {
  int n = col0 + (lane & 15);
  int mofs = (lane >> 4) * 8;
#pragma unroll
  for (int j = 0; j < 8; ++j) {
    int m = row0 + j + mofs;        // m = b*T + t
    int b = m >> 11, t = m & (TLEN - 1);
    float s = siluf(acc[j]);
    if (n < DINN) u[(size_t)t * (BATCH * DINN) + (size_t)b * DINN + n] = (bf16)s;
    else          gz[(size_t)t * (BATCH * DINN) + (size_t)b * DINN + (n - DINN)] = s;
  }
}

__global__ void proj_in_kernel(const bf16* __restrict__ xb, const bf16* __restrict__ winb,
                               bf16* __restrict__ u, float* __restrict__ gz) {
  const int NT = (2 * DINN) / 32;  // 64 column supertiles
  int wave = blockIdx.x * (blockDim.x >> 5) + (threadIdx.x >> 5);
  int lane = threadIdx.x & 31;
  int tiles = (NROW / 32) * NT;    // 512*64 = 32768
  if (wave >= tiles) return;
  int tm = wave / NT, tn = wave % NT;
  int row0 = tm * 32, col0 = tn * 32;

  const bf16* a0 = xb + (size_t)row0 * DIMD;
  const bf16* a1 = a0 + (size_t)16 * DIMD;
  const bf16* b0 = winb + (size_t)col0 * DIMD;
  const bf16* b1 = b0 + (size_t)16 * DIMD;
  v8f c00 = acc_zero(), c01 = acc_zero(), c10 = acc_zero(), c11 = acc_zero();
#pragma unroll 2
  for (int k = 0; k < DIMD; k += 32) {
    v16bf av0 = load_a(a0 + k, DIMD, lane);
    v16bf av1 = load_a(a1 + k, DIMD, lane);
    v16bf bv0 = load_b(b0 + k, DIMD, lane);
    v16bf bv1 = load_b(b1 + k, DIMD, lane);
    c00 = wmma_bf16(av0, bv0, c00);
    c01 = wmma_bf16(av0, bv1, c01);
    c10 = wmma_bf16(av1, bv0, c10);
    c11 = wmma_bf16(av1, bv1, c11);
  }
  store_xz_tile(c00, row0,      col0,      lane, u, gz);
  store_xz_tile(c01, row0,      col0 + 16, lane, u, gz);
  store_xz_tile(c10, row0 + 16, col0,      lane, u, gz);
  store_xz_tile(c11, row0 + 16, col0 + 16, lane, u, gz);
}

// ---------------------------------------------------------------------------
// Persistent recurrence kernel: 16 blocks x 256 threads = 128 waves.
//   phase 1 (128 waves, one 16-col d_state tile each, wave-private columns):
//       h_new = tanh(sigmoid(log_A)*h + u_t @ B); h16 is double-buffered.
//   single grid barrier
//   phase 2 (all 128 waves: wave pairs split K=2048 on one d_inner tile,
//            partials reduced through LDS):
//       y_t = (h_new @ C) * gz_t
// ---------------------------------------------------------------------------
__global__ void recurrence_kernel(const bf16* __restrict__ u, const float* __restrict__ gz,
                                  const bf16* __restrict__ btb, const bf16* __restrict__ ctb,
                                  const float* __restrict__ logA, float* __restrict__ h32,
                                  bf16* __restrict__ h16, bf16* __restrict__ yb,
                                  float* __restrict__ hfinal, unsigned* __restrict__ sync) {
  __shared__ float red[8 * 256];   // 8 waves/block x (32 lanes x v8f)

  int lane = threadIdx.x & 31;
  int lwave = threadIdx.x >> 5;                                   // 0..7
  int wave = blockIdx.x * (blockDim.x >> 5) + lwave;              // 0..127
  int mofs = (lane >> 4) * 8;
  long long gid = (long long)blockIdx.x * blockDim.x + threadIdx.x;  // 0..4095

  // phase-1 per-lane state column (constant across t)
  int n1 = wave * 16 + (lane & 15);                  // 0..2047
  float a_sig = 1.0f / (1.0f + __expf(-logA[n1]));
  const bf16* brow1 = btb + (size_t)(wave * 16) * DINN;           // Bt [n][k]

  // phase-2 tiling: pair (2w, 2w+1) shares y tile w, splits K halves
  int tile2 = wave >> 1;                                          // 0..63
  int kh = wave & 1;
  const bf16* brow2 = ctb + (size_t)(tile2 * 16) * DST + kh * 1024;  // Ct [n][k]

  for (int t = 0; t < TLEN; ++t) {
    const bf16* arow = u + (size_t)t * (BATCH * DINN);
    bf16* hbuf = h16 + (size_t)(t & 1) * 16 * DST;

    // ---- phase 1: h update (wave-private columns) ------------------------
    {
      v8f acc = acc_zero();
#pragma unroll 4
      for (int k = 0; k < DINN; k += 32) {
        v16bf av = load_a_pad8(arow + k, DINN, lane);
        v16bf bv = load_b(brow1 + k, DINN, lane);
        acc = wmma_bf16(av, bv, acc);
      }
#pragma unroll
      for (int j = 0; j < 8; ++j) {
        int m = j + mofs;
        size_t idx = (size_t)m * DST + n1;
        float hn = tanhf(a_sig * h32[idx] + acc[j]);  // rows>=8 stay 0
        h32[idx] = hn;
        hbuf[idx] = (bf16)hn;
      }
      // prefetch next step's activations while we wait at the barrier
      if (t + 1 < TLEN) {
        __builtin_prefetch(u + (size_t)(t + 1) * (BATCH * DINN) + gid * 2, 0, 1);
        __builtin_prefetch(gz + (size_t)(t + 1) * (BATCH * DINN) + gid * 2, 0, 1);
      }
    }
    grid_barrier(&sync[0], &sync[1]);

    // ---- phase 2: y = (h @ C) * silu(z), K split across wave pairs -------
    {
      const bf16* arow2 = hbuf + kh * 1024;
      v8f acc = acc_zero();
#pragma unroll 4
      for (int k = 0; k < 1024; k += 32) {
        v16bf av = load_a(arow2 + k, DST, lane);
        v16bf bv = load_b(brow2 + k, DST, lane);
        acc = wmma_bf16(av, bv, acc);
      }
      float* myred = &red[lwave * 256 + lane * 8];
#pragma unroll
      for (int j = 0; j < 8; ++j) myred[j] = acc[j];
      __syncthreads();
      if (kh == 0) {
        const float* pr = &red[(lwave + 1) * 256 + lane * 8];
        int n = tile2 * 16 + (lane & 15);            // 0..1023
#pragma unroll
        for (int j = 0; j < 8; ++j) {
          int b = j + mofs;
          if (b < BATCH) {
            float v = acc[j] + pr[j];
            float g = gz[(size_t)t * (BATCH * DINN) + (size_t)b * DINN + n];
            yb[((size_t)b * TLEN + t) * DINN + n] = (bf16)(v * g);
          }
        }
      }
    }
    // next write to `red` happens after the next grid_barrier's __syncthreads
  }

  // h_final (second output of the tuple): rows 0..7 of h32.
  long long nthr = (long long)gridDim.x * blockDim.x;
  for (long long i = gid; i < (long long)BATCH * DST; i += nthr) hfinal[i] = h32[i];
}

// ---------------------------------------------------------------------------
// Output projection: out = y @ W_out^T (f32 into d_out), 2x2 register blocking
// ---------------------------------------------------------------------------
__global__ void proj_out_kernel(const bf16* __restrict__ yb, const bf16* __restrict__ woutb,
                                float* __restrict__ out) {
  const int NT = DIMD / 32;  // 32 column supertiles
  int wave = blockIdx.x * (blockDim.x >> 5) + (threadIdx.x >> 5);
  int lane = threadIdx.x & 31;
  int tiles = (NROW / 32) * NT;    // 512*32 = 16384
  if (wave >= tiles) return;
  int tm = wave / NT, tn = wave % NT;
  int row0 = tm * 32, col0 = tn * 32;

  const bf16* a0 = yb + (size_t)row0 * DINN;
  const bf16* a1 = a0 + (size_t)16 * DINN;
  const bf16* b0 = woutb + (size_t)col0 * DINN;
  const bf16* b1 = b0 + (size_t)16 * DINN;
  v8f c00 = acc_zero(), c01 = acc_zero(), c10 = acc_zero(), c11 = acc_zero();
#pragma unroll 2
  for (int k = 0; k < DINN; k += 32) {
    v16bf av0 = load_a(a0 + k, DINN, lane);
    v16bf av1 = load_a(a1 + k, DINN, lane);
    v16bf bv0 = load_b(b0 + k, DINN, lane);
    v16bf bv1 = load_b(b1 + k, DINN, lane);
    c00 = wmma_bf16(av0, bv0, c00);
    c01 = wmma_bf16(av0, bv1, c01);
    c10 = wmma_bf16(av1, bv0, c10);
    c11 = wmma_bf16(av1, bv1, c11);
  }
  int nc = lane & 15;
  int mofs = (lane >> 4) * 8;
#pragma unroll
  for (int j = 0; j < 8; ++j) {
    int m0 = row0 + j + mofs;
    out[(size_t)m0 * DIMD + col0 + nc]           = c00[j];
    out[(size_t)m0 * DIMD + col0 + 16 + nc]      = c01[j];
    out[(size_t)(m0 + 16) * DIMD + col0 + nc]    = c10[j];
    out[(size_t)(m0 + 16) * DIMD + col0 + 16 + nc] = c11[j];
  }
}

// ---------------------------------------------------------------------------
// Host launcher
// ---------------------------------------------------------------------------
extern "C" void kernel_launch(void* const* d_in, const int* in_sizes, int n_in,
                              void* d_out, int out_size, void* d_ws, size_t ws_size,
                              hipStream_t stream) {
  (void)in_sizes; (void)n_in; (void)out_size; (void)ws_size;
  const float* x     = (const float*)d_in[0];
  const float* h0    = (const float*)d_in[1];
  const float* W_in  = (const float*)d_in[2];
  const float* W_out = (const float*)d_in[3];
  const float* B_mat = (const float*)d_in[4];
  const float* C_mat = (const float*)d_in[5];
  const float* log_A = (const float*)d_in[6];
  float* out = (float*)d_out;

  size_t off = 0;
  char* ws = (char*)d_ws;
  auto take = [&](size_t bytes) {
    void* p = ws + off;
    off += (bytes + 255) & ~(size_t)255;
    return p;
  };
  bf16*     u     = (bf16*)take((size_t)TLEN * BATCH * DINN * 2);      // 32 MB
  float*    gz    = (float*)take((size_t)TLEN * BATCH * DINN * 4);     // 64 MB
  bf16*     yb    = (bf16*)take((size_t)NROW * DINN * 2);              // 32 MB
  bf16*     xb    = (bf16*)take((size_t)NROW * DIMD * 2);              // 32 MB
  bf16*     winb  = (bf16*)take((size_t)2 * DINN * DIMD * 2);          //  4 MB
  bf16*     woutb = (bf16*)take((size_t)DIMD * DINN * 2);              //  2 MB
  bf16*     btb   = (bf16*)take((size_t)DST * DINN * 2);               //  4 MB (Bt[n][k])
  bf16*     ctb   = (bf16*)take((size_t)DINN * DST * 2);               //  4 MB (Ct[n][k])
  float*    h32   = (float*)take((size_t)16 * DST * 4);
  bf16*     h16   = (bf16*)take((size_t)2 * 16 * DST * 2);             // double buffer
  unsigned* syncb = (unsigned*)take(256);

  // 1) bf16 conversions (+ transposes for B operands)
  conv_f2b_kernel<<<4096, 256, 0, stream>>>(x, xb, (long long)NROW * DIMD);
  conv_f2b_kernel<<<2048, 256, 0, stream>>>(W_in, winb, (long long)2 * DINN * DIMD);
  conv_f2b_kernel<<<1024, 256, 0, stream>>>(W_out, woutb, (long long)DIMD * DINN);
  transpose_f2b_kernel<<<2048, 256, 0, stream>>>(B_mat, btb, DINN, DST);
  transpose_f2b_kernel<<<2048, 256, 0, stream>>>(C_mat, ctb, DST, DINN);
  init_state_kernel<<<32, 256, 0, stream>>>(h0, h32, syncb);

  // 2) input projection + SiLU split (32768 waves, 2x2 blocked)
  proj_in_kernel<<<4096, 256, 0, stream>>>(xb, winb, u, gz);

  // 3) persistent sequential scan (16 blocks = 128 waves, all co-resident)
  recurrence_kernel<<<16, 256, 0, stream>>>(u, gz, btb, ctb, log_A, h32, h16, yb,
                                            out + (size_t)NROW * DIMD, syncb);

  // 4) output projection (16384 waves, 2x2 blocked)
  proj_out_kernel<<<2048, 256, 0, stream>>>(yb, woutb, out);
}
